// GRUCausalMMM_68358699483894
// MI455X (gfx1250) — matrix operand order repaired and checked by reading, not compiled, not run on hardware
//
#include <hip/hip_runtime.h>
#include <hip/hip_bf16.h>
#include <math.h>

#define B_  256
#define T_  512
#define M_  16
#define C_  16
#define H_  768
#define H3_ 2304
#define KIN_ 32   // M + C

typedef __attribute__((ext_vector_type(16))) _Float16 v16h;
typedef __attribute__((ext_vector_type(8)))  float    v8f;

union FragU { v16h v; unsigned int u[8]; };

__device__ __forceinline__ v8f wmma_f16(v16h a, v16h b, v8f c) {
    return __builtin_amdgcn_wmma_f32_16x16x32_f16(
        /*neg_a=*/false, a, /*neg_b=*/false, b,
        /*c_mod=*/(short)0, c, /*reuse_a=*/false, /*reuse_b=*/false);
}

// A fragment (16x32 f16, M rows). lane L: row = L&15; lanes<16 hold K 0-7,16-23;
// lanes>=16 hold K 8-15,24-31. khalf = (L>>4)*8.
__device__ __forceinline__ v16h load_a_frag(const _Float16* base, int row, int ld,
                                            int kbase, int khalf) {
    FragU f;
    const _Float16* p = base + row * ld + kbase + khalf;
#pragma unroll
    for (int e = 0; e < 4; ++e) f.u[e]     = *(const unsigned int*)(p + 2 * e);
#pragma unroll
    for (int e = 0; e < 4; ++e) f.u[4 + e] = *(const unsigned int*)(p + 16 + 2 * e);
    return f.v;
}

// B fragment (32x16 f16, KxN). lane L: col = L&15; lanes<16 K 0-7,16-23, lanes>=16 K 8-15,24-31.
// w is row-major [ocol][k] (B = W.T). kb = (L>>4)*16 (offset used: kb/2).
__device__ __forceinline__ v16h load_b_frag(const _Float16* __restrict__ w, int ocol,
                                            int ld, int kbase, int kb) {
    FragU f;
    const _Float16* p = w + (size_t)ocol * ld + kbase + (kb >> 1);
#pragma unroll
    for (int e = 0; e < 4; ++e) f.u[e] = *(const unsigned int*)(p + 2 * e);
#pragma unroll
    for (int e = 0; e < 4; ++e) f.u[4 + e] = *(const unsigned int*)(p + 16 + 2 * e);
    return f.v;
}

__device__ __forceinline__ float sigmoidf_(float x) { return 1.0f / (1.0f + __expf(-x)); }
__device__ __forceinline__ float softplusf_(float x) {
    return (x > 20.0f) ? x : log1pf(__expf(x));
}

// ---------------- Kernel 1: causal encoder -> Z [B,T,M] ----------------
__global__ __launch_bounds__(256) void enc_kernel(
    const float* __restrict__ Xm, const float* __restrict__ A,
    const float* __restrict__ ew1, const float* __restrict__ eb1,
    const float* __restrict__ ew2, const float* __restrict__ eb2,
    const float* __restrict__ nw,  const float* __restrict__ nb,
    float* __restrict__ Zout) {
    __shared__ float sA[256], sW2[256], sW1[512], sNW[512];
    __shared__ float sB1[16], sB2[16], sNB[16];
    int tid = threadIdx.x;
    sA[tid] = A[tid];  sW2[tid] = ew2[tid];
    sW1[tid] = ew1[tid]; sW1[tid + 256] = ew1[tid + 256];
    sNW[tid] = nw[tid];  sNW[tid + 256] = nw[tid + 256];
    if (tid < 16) { sB1[tid] = eb1[tid]; sB2[tid] = eb2[tid]; sNB[tid] = nb[tid]; }
    __syncthreads();

    int pos = blockIdx.x * 256 + tid;              // pos = b*T + t
    float xm[16], cat[32], he1[16], he2[16];
#pragma unroll
    for (int i = 0; i < 16; ++i) xm[i] = Xm[(size_t)pos * 16 + i];
    // send[i] = sum_j xm[j]*A[i][j]; recv[i] = sum_j xm[j]*A[j][i]
#pragma unroll
    for (int i = 0; i < 16; ++i) {
        float s = 0.f, r = 0.f;
#pragma unroll
        for (int j = 0; j < 16; ++j) { s += xm[j] * sA[i * 16 + j]; r += xm[j] * sA[j * 16 + i]; }
        cat[i] = s; cat[16 + i] = r;
    }
#pragma unroll
    for (int i = 0; i < 16; ++i) {
        float a = sB1[i];
#pragma unroll
        for (int k = 0; k < 32; ++k) a += cat[k] * sW1[i * 32 + k];
        he1[i] = fmaxf(a, 0.f);
    }
#pragma unroll
    for (int i = 0; i < 16; ++i) {
        float a = sB2[i];
#pragma unroll
        for (int k = 0; k < 16; ++k) a += he1[k] * sW2[i * 16 + k];
        he2[i] = fmaxf(a, 0.f);
    }
#pragma unroll
    for (int i = 0; i < 16; ++i) {
        float a = sNB[i];
#pragma unroll
        for (int k = 0; k < 16; ++k) a += xm[k] * sNW[i * 32 + k];
#pragma unroll
        for (int k = 0; k < 16; ++k) a += he2[k] * sNW[i * 32 + 16 + k];
        Zout[(size_t)pos * 16 + i] = fmaxf(a, 0.f);
    }
}

// ------- Kernel 2: adstock scan + hill + pack GRU input (f16 [T][B][32]) -------
__global__ __launch_bounds__(256) void adstock_kernel(
    const float* __restrict__ Xm, const float* __restrict__ Xc,
    const float* __restrict__ alpha, const float* __restrict__ hill_a,
    const float* __restrict__ hill_g, const float* __restrict__ Z,
    float* __restrict__ hillw, _Float16* __restrict__ gx) {
    int idx = blockIdx.x * 256 + threadIdx.x;      // B*32 threads
    int b = idx >> 5, j = idx & 31;
    if (j < 16) {
        int m = j;
        float a  = fminf(fmaxf(alpha[m], 0.f), 1.f);
        float ha = fminf(fmaxf(hill_a[m], 0.1f), 3.0f);
        float hg = fminf(fmaxf(hill_g[m], 0.1f), 2.0f);
        float dadd = powf(hg, ha) + 1e-8f;
        const float* xrow = Xm + (size_t)b * T_ * 16 + m;
        // pass 1: running max of adstock
        float prev = 0.f, cmax = 0.f;
        for (int t = 0; t < T_; ++t) {
            float cur = xrow[(size_t)t * 16] + a * prev;
            cmax = fmaxf(cmax, cur);
            prev = cur;
        }
        cmax = fmaxf(cmax, 1e-6f);
        float inv = 1.0f / cmax;
        // pass 2: hill + pack
        prev = 0.f;
        for (int t = 0; t < T_; ++t) {
            float cur = xrow[(size_t)t * 16] + a * prev;
            prev = cur;
            float xn = fmaxf(cur * inv, 0.f);
            float num = powf(xn, ha);
            float hill = num / (num + dadd);
            size_t pi = ((size_t)b * T_ + t) * 16 + m;
            hillw[pi] = hill;
            gx[(size_t)t * B_ * 32 + (size_t)b * 32 + m] = (_Float16)(hill + Z[pi]);
        }
    } else {
        int c = j - 16;
        for (int t = 0; t < T_; ++t)
            gx[(size_t)t * B_ * 32 + (size_t)b * 32 + 16 + c] =
                (_Float16)Xc[((size_t)b * T_ + t) * 16 + c];
    }
}

// ---------------- Kernel 3: fp32 -> f16 conversion ----------------
__global__ void cvt_kernel(const float* __restrict__ src, _Float16* __restrict__ dst, int n) {
    int i = blockIdx.x * 256 + threadIdx.x;
    if (i < n) dst[i] = (_Float16)src[i];
}

// ---------------- Kernel 4: reg_term = sum(reg_emb[0]) * 0.3 ----------------
__global__ __launch_bounds__(256) void reg_kernel(const float* __restrict__ reg_emb,
                                                  float* __restrict__ out) {
    __shared__ float red[256];
    int tid = threadIdx.x;
    float s = 0.f;
    for (int i = tid; i < H_; i += 256) s += reg_emb[i];   // row 0
    red[tid] = s; __syncthreads();
    for (int w = 128; w > 0; w >>= 1) { if (tid < w) red[tid] += red[tid + w]; __syncthreads(); }
    if (tid == 0) out[0] = red[0] * 0.3f;
}

// ---------- Kernel 5: persistent GRU over time + fused w_pos/contrib head ----------
__global__ __launch_bounds__(256) void gru_kernel(
    const _Float16* __restrict__ gx,     // [T][B][32] f16
    const _Float16* __restrict__ wih,    // [3H][32]   f16
    const _Float16* __restrict__ whh,    // [3H][768]  f16
    const float* __restrict__ b_ih, const float* __restrict__ b_hh,
    const _Float16* __restrict__ wraw,   // [16][768]  f16
    const float* __restrict__ wraw_b, const float* __restrict__ h0,
    const float* __restrict__ hillw,     // [B][T][16] f32
    float* __restrict__ wpos, float* __restrict__ contrib) {
    __shared__ _Float16 hbuf[2][16 * H_];   // ping-pong h state, 48 KB
    __shared__ _Float16 xtile[16 * KIN_];   // per-step input tile (async-staged)

    const int tid  = threadIdx.x;
    const int lane = tid & 31;
    const int wave = tid >> 5;
    const int b0   = blockIdx.x * 16;
    const int rowA  = lane & 15;            // A-matrix row this lane owns
    const int khalf = (lane >> 4) * 8;      // A-frag K sub-offset
    const int kb    = (lane >> 4) * 16;     // B-frag K sub-offset
    const int ncol  = lane & 15;            // B/D column this lane owns

    // this thread's slot in the x tile (one dword each): row r, even k
    const int xr = (tid * 2) >> 5;
    const int xk = (tid * 2) & 31;
    const unsigned xlds = (unsigned)(unsigned long long)(void*)&xtile[xr * 32 + xk];
    const unsigned long long xgbase =
        (unsigned long long)(const void*)&gx[(size_t)(b0 + xr) * 32 + xk];

    // init h from h0 (broadcast over batch rows)
    for (int i = tid; i < 16 * H_; i += 256) hbuf[0][i] = (_Float16)h0[i % H_];

    // per-tile biases (6 hidden tiles of 16 per wave), constant over t
    float bR[6], bZ[6], bI[6], bHn[6];
#pragma unroll
    for (int i = 0; i < 6; ++i) {
        int j = (wave * 6 + i) * 16 + ncol;
        bR[i]  = b_ih[j] + b_hh[j];
        bZ[i]  = b_ih[H_ + j] + b_hh[H_ + j];
        bI[i]  = b_ih[2 * H_ + j];
        bHn[i] = b_hh[2 * H_ + j];
    }
    const float wb = wraw_b[ncol];
    __syncthreads();

    for (int t = 0; t < T_; ++t) {
        _Float16* cur = hbuf[t & 1];
        _Float16* nxt = hbuf[(t & 1) ^ 1];

        // stage x_t tile [16 x 32] straight into LDS via async DMA (ASYNCcnt path)
        {
            unsigned long long ga = xgbase + (unsigned long long)t * (B_ * 32 * 2);
            asm volatile("global_load_async_to_lds_b32 %0, %1, off"
                         :: "v"(xlds), "v"(ga) : "memory");
            asm volatile("s_wait_asynccnt 0x0" ::: "memory");
        }
        // prefetch next step's x tile region (global_prefetch_b8)
        if (wave == 1 && t + 1 < T_) {
            __builtin_prefetch(
                (const char*)gx + ((size_t)(t + 1) * B_ * 32 + (size_t)b0 * 32 + lane * 16) * 2,
                0, 0);
        }
        __syncthreads();

#pragma unroll 1
        for (int i = 0; i < 6; ++i) {
            const int jt = wave * 6 + i, j0 = jt * 16;
            v8f ar = {}, az = {}, ain = {}, ahn = {};
            // gh = h @ W_hh.T  (K = 768)
#pragma unroll 1
            for (int kt = 0; kt < H_ / 32; ++kt) {
                v16h a  = load_a_frag(cur, rowA, H_, kt * 32, khalf);
                v16h br = load_b_frag(whh, j0 + ncol,          H_, kt * 32, kb);
                v16h bz = load_b_frag(whh, H_ + j0 + ncol,     H_, kt * 32, kb);
                v16h bn = load_b_frag(whh, 2 * H_ + j0 + ncol, H_, kt * 32, kb);
                ar  = wmma_f16(a, br, ar);
                az  = wmma_f16(a, bz, az);
                ahn = wmma_f16(a, bn, ahn);
            }
            // gi = x @ W_ih.T  (K = 32)
            {
                v16h ax = load_a_frag(xtile, rowA, KIN_, 0, khalf);
                ar  = wmma_f16(ax, load_b_frag(wih, j0 + ncol,          KIN_, 0, kb), ar);
                az  = wmma_f16(ax, load_b_frag(wih, H_ + j0 + ncol,     KIN_, 0, kb), az);
                ain = wmma_f16(ax, load_b_frag(wih, 2 * H_ + j0 + ncol, KIN_, 0, kb), ain);
            }
            // gates + state update; D element (e,lane) -> row = e + 8*(lane>>4), col = ncol
            const int j = j0 + ncol;
#pragma unroll
            for (int e = 0; e < 8; ++e) {
                int row = e + 8 * (lane >> 4);
                float hprev = (float)cur[row * H_ + j];
                float r = sigmoidf_(ar[e] + bR[i]);
                float z = sigmoidf_(az[e] + bZ[i]);
                float n = tanhf(ain[e] + bI[i] + r * (ahn[e] + bHn[i]));
                nxt[row * H_ + j] = (_Float16)((1.0f - z) * n + z * hprev);
            }
        }
        __syncthreads();   // nxt complete

        // fused head: w_pos = softplus(h_new @ wraw_w.T + b); contrib = hill * w_pos
        if (wave == 0) {
            v8f acc = {};
#pragma unroll 1
            for (int kt = 0; kt < H_ / 32; ++kt) {
                v16h a = load_a_frag(nxt, rowA, H_, kt * 32, khalf);
                v16h b = load_b_frag(wraw, ncol, H_, kt * 32, kb);
                acc = wmma_f16(a, b, acc);
            }
#pragma unroll
            for (int e = 0; e < 8; ++e) {
                int row = e + 8 * (lane >> 4);
                float wp = softplusf_(acc[e] + wb);
                size_t oi = ((size_t)(b0 + row) * T_ + t) * 16 + ncol;
                wpos[oi] = wp;
                contrib[oi] = hillw[oi] * wp;
            }
        }
    }
}

// -------- Kernel 6: t=0 fix-up: w0 = 0.5*wp[:,1] + 0.5*wp[:,0] --------
__global__ void headfix_kernel(float* __restrict__ wpos, float* __restrict__ contrib,
                               const float* __restrict__ hillw) {
    int idx = blockIdx.x * 256 + threadIdx.x;     // B*16 threads
    int b = idx >> 4, m = idx & 15;
    size_t i0 = ((size_t)b * T_ + 0) * 16 + m;
    size_t i1 = ((size_t)b * T_ + 1) * 16 + m;
    float w0 = 0.5f * wpos[i1] + 0.5f * wpos[i0];
    wpos[i0] = w0;
    contrib[i0] = hillw[i0] * w0;
}

// -------- Kernel 7: y = sum_m contrib + 0.3*ctrl + reg + bias --------
__global__ __launch_bounds__(256) void y_kernel(
    const float* __restrict__ contrib, const float* __restrict__ Xc,
    const float* __restrict__ ctrl_w, const float* __restrict__ ctrl_b,
    const float* __restrict__ regv, const float* __restrict__ bias,
    float* __restrict__ y) {
    __shared__ float scw[H_ * 16];   // 48 KB
    __shared__ float scb[H_];
    int tid = threadIdx.x;
    for (int i = tid; i < H_ * 16; i += 256) scw[i] = ctrl_w[i];
    for (int i = tid; i < H_; i += 256) scb[i] = ctrl_b[i];
    __syncthreads();

    size_t pos = (size_t)blockIdx.x * 256 + tid;   // b*T + t
    float msum = 0.f;
#pragma unroll
    for (int m = 0; m < 16; ++m) msum += contrib[pos * 16 + m];
    float cx[16];
#pragma unroll
    for (int c = 0; c < 16; ++c) cx[c] = Xc[pos * 16 + c];
    float csum = 0.f;
    for (int o = 0; o < H_; ++o) {
        float a = scb[o];
#pragma unroll
        for (int c = 0; c < 16; ++c) a += cx[c] * scw[o * 16 + c];
        csum += fmaxf(a, 0.f);
    }
    y[pos] = msum + 0.3f * csum + regv[0] + bias[0];
}

extern "C" void kernel_launch(void* const* d_in, const int* in_sizes, int n_in,
                              void* d_out, int out_size, void* d_ws, size_t ws_size,
                              hipStream_t stream) {
    const float* Xm     = (const float*)d_in[0];
    const float* Xc     = (const float*)d_in[1];
    /* d_in[2] = R (unused in forward) */
    const float* A      = (const float*)d_in[3];
    const float* ew1    = (const float*)d_in[4];
    const float* eb1    = (const float*)d_in[5];
    const float* ew2    = (const float*)d_in[6];
    const float* eb2    = (const float*)d_in[7];
    const float* nw     = (const float*)d_in[8];
    const float* nb     = (const float*)d_in[9];
    const float* Wih    = (const float*)d_in[10];
    const float* Whh    = (const float*)d_in[11];
    const float* bih    = (const float*)d_in[12];
    const float* bhh    = (const float*)d_in[13];
    const float* wrw    = (const float*)d_in[14];
    const float* wrb    = (const float*)d_in[15];
    const float* alpha  = (const float*)d_in[16];
    const float* hill_a = (const float*)d_in[17];
    const float* hill_g = (const float*)d_in[18];
    const float* ctw    = (const float*)d_in[19];
    const float* ctb    = (const float*)d_in[20];
    const float* regemb = (const float*)d_in[21];
    const float* bias   = (const float*)d_in[22];
    const float* h0     = (const float*)d_in[23];

    float* y       = (float*)d_out;                       // [B*T]
    float* wpos    = y + (size_t)B_ * T_;                 // [B*T*M]
    float* contrib = wpos + (size_t)B_ * T_ * M_;         // [B*T*M]

    // workspace partition (all offsets 256B aligned by construction)
    char* w = (char*)d_ws;
    float*    Zw     = (float*)w;            w += (size_t)B_ * T_ * M_ * 4;   // 8 MB
    float*    hillw  = (float*)w;            w += (size_t)B_ * T_ * M_ * 4;   // 8 MB
    _Float16* gx     = (_Float16*)w;         w += (size_t)T_ * B_ * 32 * 2;   // 8 MB
    _Float16* wih16  = (_Float16*)w;         w += (size_t)H3_ * 32 * 2;
    _Float16* whh16  = (_Float16*)w;         w += (size_t)H3_ * H_ * 2;
    _Float16* wraw16 = (_Float16*)w;         w += (size_t)M_ * H_ * 2;
    float*    regv   = (float*)w;            w += 256;

    enc_kernel<<<(B_ * T_) / 256, 256, 0, stream>>>(Xm, A, ew1, eb1, ew2, eb2, nw, nb, Zw);
    adstock_kernel<<<(B_ * 32) / 256, 256, 0, stream>>>(Xm, Xc, alpha, hill_a, hill_g,
                                                        Zw, hillw, gx);
    cvt_kernel<<<(H3_ * 32 + 255) / 256, 256, 0, stream>>>(Wih, wih16, H3_ * 32);
    cvt_kernel<<<(H3_ * H_ + 255) / 256, 256, 0, stream>>>(Whh, whh16, H3_ * H_);
    cvt_kernel<<<(M_ * H_ + 255) / 256, 256, 0, stream>>>(wrw, wraw16, M_ * H_);
    reg_kernel<<<1, 256, 0, stream>>>(regemb, regv);
    gru_kernel<<<B_ / 16, 256, 0, stream>>>(gx, wih16, whh16, bih, bhh, wraw16, wrb, h0,
                                            hillw, wpos, contrib);
    headfix_kernel<<<(B_ * 16) / 256, 256, 0, stream>>>(wpos, contrib, hillw);
    y_kernel<<<(B_ * T_) / 256, 256, 0, stream>>>(contrib, Xc, ctw, ctb, regv, bias, y);
}
